// TemporalEGCNEncoder_55722905698996
// MI455X (gfx1250) — matrix operand order repaired and checked by reading, not compiled, not run on hardware
//
#include <hip/hip_runtime.h>

// Temporal EGCN encoder for MI455X (gfx1250), bf16 WMMA (V_WMMA_F32_16X16X32_BF16).
// ~180MB mandatory HBM traffic @ 23.3TB/s; per-step A (2MB bf16) is L2-resident
// across its reuses. Single-wave (wave32) workgroups for WGP spread; native
// f32->bf16 conversions to keep the conversion kernel bandwidth-bound.

#define DEVI __device__ __forceinline__

typedef __bf16 bf16;
typedef __attribute__((ext_vector_type(4)))  __bf16 v4bf;
typedef __attribute__((ext_vector_type(8)))  __bf16 v8bf;
typedef __attribute__((ext_vector_type(16))) __bf16 v16bf;
typedef __attribute__((ext_vector_type(4)))  float  v4f;
typedef __attribute__((ext_vector_type(8)))  float  v8f;

constexpr int B_ = 4, T_ = 24, N_ = 512, F_ = 64, DIN = 128;

DEVI bf16 f2bf(float f) { return (bf16)f; }   // native v_cvt path on gfx1250

DEVI float sigmoidf_(float x) { return 1.0f / (1.0f + __expf(-x)); }

// ---- WMMA fragment helpers (contiguous-K order, same for A and B) ----------------------------
// A-frag 16x32 bf16, row-major source with row stride ld:
//   lane<16 : row=lane,    K = {0..15}
//   lane>=16: row=lane-16, K = {16..31}
DEVI v16bf load_a(const bf16* p, int ld, int lane) {
  int r  = lane & 15;
  int kb = (lane >> 4) << 4;
  const bf16* rp = p + (size_t)r * ld + kb;
  v8bf lo = *(const v8bf*)rp;
  v8bf hi = *(const v8bf*)(rp + 8);
  v16bf a;
#pragma unroll
  for (int i = 0; i < 8; ++i) { a[i] = lo[i]; a[i + 8] = hi[i]; }
  return a;
}

// Graph GEMM: acc += A_tile[16 x 512] * V[512 x 16*NCT], with V supplied
// feature-major (transposed) so each B-frag is one contiguous 32B lane load:
//   lane col = lane&15, K base = (lane<16 ? 0 : 16), 16 contiguous K per lane.
template <int NCT, class FP>
DEVI void gemm_graph(v8f (&acc)[NCT], const bf16* Atile, FP featRow, int lane) {
  int col_l = lane & 15;
  int koff  = (lane >> 4) << 4;
  int r = lane & 15, kb = (lane >> 4) << 4;
  const bf16* arow = Atile + (size_t)r * N_ + kb;
  for (int kc = 0; kc < N_ / 32; ++kc) {
    v8bf lo = *(const v8bf*)(arow + kc * 32);
    v8bf hi = *(const v8bf*)(arow + kc * 32 + 8);
    v16bf a;
#pragma unroll
    for (int i = 0; i < 8; ++i) { a[i] = lo[i]; a[i + 8] = hi[i]; }
    if (kc + 4 < N_ / 32) __builtin_prefetch(arow + (kc + 4) * 32, 0, 0);
#pragma unroll
    for (int ct = 0; ct < NCT; ++ct) {
      v16bf b = *(const v16bf*)(featRow(ct * 16 + col_l) + kc * 32 + koff);
      acc[ct] = __builtin_amdgcn_wmma_f32_16x16x32_bf16(
          false, a, false, b, (short)0, acc[ct], false, false);
    }
  }
}

// Dense GEMM: acc += L[16 x 32*kchunks] * W, W transposed with row stride ldw.
template <int NCT>
DEVI void gemm_dense(v8f (&acc)[NCT], const bf16* L, int ldl,
                     const bf16* WT, int ldw, int kchunks, int lane) {
  int col_l = lane & 15;
  int koff  = (lane >> 4) << 4;
  for (int kc = 0; kc < kchunks; ++kc) {
    v16bf a = load_a(L + kc * 32, ldl, lane);
#pragma unroll
    for (int ct = 0; ct < NCT; ++ct) {
      v16bf b = *(const v16bf*)(WT + (size_t)(ct * 16 + col_l) * ldw + kc * 32 + koff);
      acc[ct] = __builtin_amdgcn_wmma_f32_16x16x32_bf16(
          false, a, false, b, (short)0, acc[ct], false, false);
    }
  }
}

// ---- prep kernels ----------------------------------------------------------
// Vectorized fp32 -> bf16 conversion of the edge tensor (bandwidth-bound:
// 100MB read + 50MB write).
__global__ void k_convert_A(const float* __restrict__ e, bf16* __restrict__ Abf, long n4) {
  long i = (long)blockIdx.x * blockDim.x + threadIdx.x;
  long s = (long)gridDim.x * blockDim.x;
  for (; i < n4; i += s) {
    v4f v = *(const v4f*)(e + i * 4);
    v4bf o;
#pragma unroll
    for (int j = 0; j < 4; ++j) o[j] = f2bf(v[j]);
    *(v4bf*)(Abf + i * 4) = o;
  }
}

__global__ void k_prep_w(const float* fc_w,
                         const float* Wz1, const float* Wr1, const float* Wc1,
                         const float* Wz2, const float* Wr2, const float* Wc2,
                         bf16* fcT, bf16* WzT1, bf16* WrT1, bf16* WcT1,
                         bf16* WzT2, bf16* WrT2, bf16* WcT2) {
  for (int i = threadIdx.x; i < 64 * 64; i += blockDim.x) {
    int o = i >> 6, k = i & 63;
    fcT[o * 64 + k]  = f2bf(fc_w[k * 64 + o]);
    WzT2[o * 64 + k] = f2bf(Wz2[k * 64 + o]);
    WrT2[o * 64 + k] = f2bf(Wr2[k * 64 + o]);
    WcT2[o * 64 + k] = f2bf(Wc2[k * 64 + o]);
  }
  for (int i = threadIdx.x; i < 64 * 128; i += blockDim.x) {
    int o = i >> 7, k = i & 127;
    WzT1[o * 128 + k] = f2bf(Wz1[k * 64 + o]);
    WrT1[o * 128 + k] = f2bf(Wr1[k * 64 + o]);
    WcT1[o * 128 + k] = f2bf(Wc1[k * 64 + o]);
  }
}

__global__ void k_zero_h(float* h, bf16* hT) {
  int n = B_ * N_ * F_;
  for (int i = blockIdx.x * blockDim.x + threadIdx.x; i < n; i += gridDim.x * blockDim.x) {
    h[i] = 0.0f;
    hT[i] = f2bf(0.0f);
  }
}

// xpT[b,t,f,n] = relu(x[b,t,n,:] @ fc_w + fc_b)[f], stored feature-major bf16.
// One wave per 16-row tile.
__global__ void k_proj(const float* __restrict__ x, const float* __restrict__ fc_b,
                       const bf16* __restrict__ fcT, bf16* __restrict__ xpT) {
  int bt = blockIdx.x >> 5, rblk = blockIdx.x & 31;
  int lane = threadIdx.x & 31;
  int row0 = rblk * 16;
  const float* xrow = x + (size_t)bt * N_ * F_ + (size_t)row0 * F_;
  bf16* outT = xpT + (size_t)bt * F_ * N_;
  int col_l = lane & 15, hi8 = (lane >> 4) << 3, koff = (lane >> 4) << 4;
  v8f acc[4] = {};
  for (int kc = 0; kc < 2; ++kc) {
    v16bf a;
    {
      int r = lane & 15, kb = (lane >> 4) << 4;
      const float* p = xrow + r * F_ + kc * 32 + kb;
      v4f p0 = *(const v4f*)(p);
      v4f p1 = *(const v4f*)(p + 4);
      v4f p2 = *(const v4f*)(p + 8);
      v4f p3 = *(const v4f*)(p + 12);
#pragma unroll
      for (int i = 0; i < 4; ++i) {
        a[i]      = f2bf(p0[i]);
        a[i + 4]  = f2bf(p1[i]);
        a[i + 8]  = f2bf(p2[i]);
        a[i + 12] = f2bf(p3[i]);
      }
    }
#pragma unroll
    for (int ct = 0; ct < 4; ++ct) {
      v16bf b = *(const v16bf*)(fcT + (size_t)(ct * 16 + col_l) * 64 + kc * 32 + koff);
      acc[ct] = __builtin_amdgcn_wmma_f32_16x16x32_bf16(
          false, a, false, b, (short)0, acc[ct], false, false);
    }
  }
#pragma unroll
  for (int ct = 0; ct < 4; ++ct) {
    int col = ct * 16 + col_l;
    float bias = fc_b[col];
#pragma unroll
    for (int g = 0; g < 8; ++g) {
      float v = acc[ct][g] + bias;
      v = v > 0.0f ? v : 0.0f;
      outT[(size_t)col * N_ + row0 + g + hi8] = f2bf(v);
    }
  }
}

// ---- per-step phase kernels (single-wave workgroups, 16 rows per wave) -----
// Phase 1: Axh = A @ [xt|h]; keep Axt; hz1=relu(Axh@Wz1), hr1=relu(Axh@Wr1)
__global__ void k_phase1(int t, const bf16* __restrict__ Abf, const bf16* __restrict__ xpT,
                         const bf16* __restrict__ hT,
                         const bf16* __restrict__ WzT1, const bf16* __restrict__ WrT1,
                         bf16* __restrict__ AxtG, bf16* __restrict__ hzr1T) {
  __shared__ __align__(32) bf16 st[16 * DIN];
  int b = blockIdx.x >> 5, rblk = blockIdx.x & 31;
  int lane = threadIdx.x & 31;
  int row0 = rblk * 16;
  const bf16* Atile  = Abf + ((size_t)(b * T_ + t) * N_ + row0) * N_;
  const bf16* xpT_bt = xpT + (size_t)(b * T_ + t) * F_ * N_;
  const bf16* hT_b   = hT + (size_t)b * F_ * N_;
  int col_l = lane & 15, hi8 = (lane >> 4) << 3;

  v8f acc[8] = {};
  gemm_graph<8>(acc, Atile,
      [&](int f) -> const bf16* {
        return f < F_ ? xpT_bt + (size_t)f * N_ : hT_b + (size_t)(f - F_) * N_;
      }, lane);

  // Axt = A@xt (cols 0..63), row-major bf16, reused by the c-gate
  bf16* axt = AxtG + ((size_t)b * N_ + row0) * F_;
#pragma unroll
  for (int ct = 0; ct < 4; ++ct)
#pragma unroll
    for (int g = 0; g < 8; ++g)
      axt[(g + hi8) * F_ + ct * 16 + col_l] = f2bf(acc[ct][g]);

  // stage full Axh [16 x 128] in LDS for the weight GEMMs
#pragma unroll
  for (int ct = 0; ct < 8; ++ct)
#pragma unroll
    for (int g = 0; g < 8; ++g)
      st[(g + hi8) * DIN + ct * 16 + col_l] = f2bf(acc[ct][g]);
  __syncthreads();   // single-wave WG: S_NOP barrier, acts as LDS fence

  bf16* outT = hzr1T + (size_t)b * DIN * N_;
  {
    v8f a2[4] = {};
    gemm_dense<4>(a2, st, DIN, WzT1, DIN, 4, lane);
#pragma unroll
    for (int ct = 0; ct < 4; ++ct)
#pragma unroll
      for (int g = 0; g < 8; ++g) {
        float v = a2[ct][g]; v = v > 0.0f ? v : 0.0f;
        outT[(size_t)(ct * 16 + col_l) * N_ + row0 + g + hi8] = f2bf(v);
      }
  }
  {
    v8f a2[4] = {};
    gemm_dense<4>(a2, st, DIN, WrT1, DIN, 4, lane);
#pragma unroll
    for (int ct = 0; ct < 4; ++ct)
#pragma unroll
      for (int g = 0; g < 8; ++g) {
        float v = a2[ct][g]; v = v > 0.0f ? v : 0.0f;
        outT[(size_t)(F_ + ct * 16 + col_l) * N_ + row0 + g + hi8] = f2bf(v);
      }
  }
}

// Phase 2: Ah = A @ [hz1|hr1]; z=sigmoid(Ah_z@Wz2); r=sigmoid(Ah_r@Wr2); rh=r*h
__global__ void k_phase2(int t, const bf16* __restrict__ Abf, const bf16* __restrict__ hzr1T,
                         const bf16* __restrict__ WzT2, const bf16* __restrict__ WrT2,
                         const float* __restrict__ h, float* __restrict__ zbuf,
                         bf16* __restrict__ rhT) {
  __shared__ __align__(32) bf16 st[16 * DIN];
  int b = blockIdx.x >> 5, rblk = blockIdx.x & 31;
  int lane = threadIdx.x & 31;
  int row0 = rblk * 16;
  const bf16* Atile = Abf + ((size_t)(b * T_ + t) * N_ + row0) * N_;
  const bf16* featT = hzr1T + (size_t)b * DIN * N_;
  int col_l = lane & 15, hi8 = (lane >> 4) << 3;

  v8f acc[8] = {};
  gemm_graph<8>(acc, Atile, [&](int f) { return featT + (size_t)f * N_; }, lane);

#pragma unroll
  for (int ct = 0; ct < 8; ++ct)
#pragma unroll
    for (int g = 0; g < 8; ++g)
      st[(g + hi8) * DIN + ct * 16 + col_l] = f2bf(acc[ct][g]);
  __syncthreads();

  v8f az[4] = {}, ar[4] = {};
  gemm_dense<4>(az, st, DIN, WzT2, 64, 2, lane);        // cols 0..63 of Ah
  gemm_dense<4>(ar, st + 64, DIN, WrT2, 64, 2, lane);   // cols 64..127 of Ah

  float* zb = zbuf + ((size_t)b * N_ + row0) * F_;
  const float* hb = h + ((size_t)b * N_ + row0) * F_;
  bf16* rT = rhT + (size_t)b * F_ * N_;
#pragma unroll
  for (int ct = 0; ct < 4; ++ct)
#pragma unroll
    for (int g = 0; g < 8; ++g) {
      int col = ct * 16 + col_l, row = g + hi8;
      float z = sigmoidf_(az[ct][g]);
      zb[row * F_ + col] = z;
      float r = sigmoidf_(ar[ct][g]);
      rT[(size_t)col * N_ + row0 + row] = f2bf(r * hb[row * F_ + col]);
    }
}

// Phase 3: Arh = A @ (r*h); hc1 = relu([Axt | Arh] @ Wc1)
__global__ void k_phase3(int t, const bf16* __restrict__ Abf, const bf16* __restrict__ rhT,
                         const bf16* __restrict__ AxtG, const bf16* __restrict__ WcT1,
                         bf16* __restrict__ hc1T) {
  __shared__ __align__(32) bf16 st[16 * F_];
  int b = blockIdx.x >> 5, rblk = blockIdx.x & 31;
  int lane = threadIdx.x & 31;
  int row0 = rblk * 16;
  const bf16* Atile = Abf + ((size_t)(b * T_ + t) * N_ + row0) * N_;
  const bf16* rT = rhT + (size_t)b * F_ * N_;
  int col_l = lane & 15, hi8 = (lane >> 4) << 3;

  v8f acc[4] = {};
  gemm_graph<4>(acc, Atile, [&](int f) { return rT + (size_t)f * N_; }, lane);

#pragma unroll
  for (int ct = 0; ct < 4; ++ct)
#pragma unroll
    for (int g = 0; g < 8; ++g)
      st[(g + hi8) * F_ + ct * 16 + col_l] = f2bf(acc[ct][g]);
  __syncthreads();

  v8f a2[4] = {};
  const bf16* axt = AxtG + ((size_t)b * N_ + row0) * F_;
  gemm_dense<4>(a2, axt, F_, WcT1, DIN, 2, lane);       // K 0..63   (A@xt)
  gemm_dense<4>(a2, st, F_, WcT1 + 64, DIN, 2, lane);   // K 64..127 (A@rh)

  bf16* outT = hc1T + (size_t)b * F_ * N_;
#pragma unroll
  for (int ct = 0; ct < 4; ++ct)
#pragma unroll
    for (int g = 0; g < 8; ++g) {
      float v = a2[ct][g]; v = v > 0.0f ? v : 0.0f;
      outT[(size_t)(ct * 16 + col_l) * N_ + row0 + g + hi8] = f2bf(v);
    }
}

// Phase 4: c = tanh((A@hc1)@Wc2); hn = z*h + (1-z)*c; emit h, hT, out[:,t]
__global__ void k_phase4(int t, const bf16* __restrict__ Abf, const bf16* __restrict__ hc1T,
                         const bf16* __restrict__ WcT2, const float* __restrict__ zbuf,
                         float* __restrict__ h, bf16* __restrict__ hT,
                         float* __restrict__ out) {
  __shared__ __align__(32) bf16 st[16 * F_];
  int b = blockIdx.x >> 5, rblk = blockIdx.x & 31;
  int lane = threadIdx.x & 31;
  int row0 = rblk * 16;
  const bf16* Atile = Abf + ((size_t)(b * T_ + t) * N_ + row0) * N_;
  const bf16* cT = hc1T + (size_t)b * F_ * N_;
  int col_l = lane & 15, hi8 = (lane >> 4) << 3;

  v8f acc[4] = {};
  gemm_graph<4>(acc, Atile, [&](int f) { return cT + (size_t)f * N_; }, lane);

#pragma unroll
  for (int ct = 0; ct < 4; ++ct)
#pragma unroll
    for (int g = 0; g < 8; ++g)
      st[(g + hi8) * F_ + ct * 16 + col_l] = f2bf(acc[ct][g]);
  __syncthreads();

  v8f a2[4] = {};
  gemm_dense<4>(a2, st, F_, WcT2, 64, 2, lane);

  const float* zb = zbuf + ((size_t)b * N_ + row0) * F_;
  float* hb = h + ((size_t)b * N_ + row0) * F_;
  bf16* hTb = hT + (size_t)b * F_ * N_;
  float* ob = out + ((size_t)(b * T_ + t) * N_ + row0) * F_;
#pragma unroll
  for (int ct = 0; ct < 4; ++ct)
#pragma unroll
    for (int g = 0; g < 8; ++g) {
      int col = ct * 16 + col_l, row = g + hi8;
      float c = tanhf(a2[ct][g]);
      float z = zb[row * F_ + col];
      float hv = hb[row * F_ + col];
      float hn = z * hv + (1.0f - z) * c;
      hb[row * F_ + col] = hn;
      hTb[(size_t)col * N_ + row0 + row] = f2bf(hn);
      ob[row * F_ + col] = hn;
    }
}

// ---- launcher --------------------------------------------------------------
extern "C" void kernel_launch(void* const* d_in, const int* in_sizes, int n_in,
                              void* d_out, int out_size, void* d_ws, size_t ws_size,
                              hipStream_t stream) {
  (void)in_sizes; (void)n_in; (void)out_size; (void)ws_size;
  const float* x    = (const float*)d_in[0];
  const float* e    = (const float*)d_in[1];
  const float* fc_w = (const float*)d_in[2];
  const float* fc_b = (const float*)d_in[3];
  const float* Wz1  = (const float*)d_in[4];
  const float* Wz2  = (const float*)d_in[5];
  const float* Wr1  = (const float*)d_in[6];
  const float* Wr2  = (const float*)d_in[7];
  const float* Wc1  = (const float*)d_in[8];
  const float* Wc2  = (const float*)d_in[9];
  float* out = (float*)d_out;

  char* ws = (char*)d_ws;
  size_t off = 0;
  auto carve = [&](size_t bytes) -> char* {
    char* p = ws + off;
    off += (bytes + 255) & ~(size_t)255;
    return p;
  };
  bf16* Abf   = (bf16*)carve((size_t)B_ * T_ * N_ * N_ * 2);   // 50.3 MB
  bf16* xpT   = (bf16*)carve((size_t)B_ * T_ * F_ * N_ * 2);   // 6.3 MB
  bf16* fcT   = (bf16*)carve(64 * 64 * 2);
  bf16* WzT1  = (bf16*)carve(64 * 128 * 2);
  bf16* WrT1  = (bf16*)carve(64 * 128 * 2);
  bf16* WcT1  = (bf16*)carve(64 * 128 * 2);
  bf16* WzT2  = (bf16*)carve(64 * 64 * 2);
  bf16* WrT2  = (bf16*)carve(64 * 64 * 2);
  bf16* WcT2  = (bf16*)carve(64 * 64 * 2);
  float* h    = (float*)carve((size_t)B_ * N_ * F_ * 4);
  bf16* hT    = (bf16*)carve((size_t)B_ * F_ * N_ * 2);
  bf16* hzr1T = (bf16*)carve((size_t)B_ * DIN * N_ * 2);
  float* zbuf = (float*)carve((size_t)B_ * N_ * F_ * 4);
  bf16* rhT   = (bf16*)carve((size_t)B_ * F_ * N_ * 2);
  bf16* AxtG  = (bf16*)carve((size_t)B_ * N_ * F_ * 2);
  bf16* hc1T  = (bf16*)carve((size_t)B_ * F_ * N_ * 2);

  k_convert_A<<<2048, 256, 0, stream>>>(e, Abf, (long)B_ * T_ * N_ * N_ / 4);
  k_prep_w<<<1, 256, 0, stream>>>(fc_w, Wz1, Wr1, Wc1, Wz2, Wr2, Wc2,
                                  fcT, WzT1, WrT1, WcT1, WzT2, WrT2, WcT2);
  k_zero_h<<<256, 256, 0, stream>>>(h, hT);
  k_proj<<<B_ * T_ * 32, 32, 0, stream>>>(x, fc_b, fcT, xpT);

  for (int t = 0; t < T_; ++t) {
    k_phase1<<<B_ * 32, 32, 0, stream>>>(t, Abf, xpT, hT, WzT1, WrT1, AxtG, hzr1T);
    k_phase2<<<B_ * 32, 32, 0, stream>>>(t, Abf, hzr1T, WzT2, WrT2, h, zbuf, rhT);
    k_phase3<<<B_ * 32, 32, 0, stream>>>(t, Abf, rhT, AxtG, WcT1, hc1T);
    k_phase4<<<B_ * 32, 32, 0, stream>>>(t, Abf, hc1T, WcT2, zbuf, h, hT, out);
  }
}
